// AdditiveAttention_86792699117973
// MI455X (gfx1250) — matrix operand order repaired and checked by reading, not compiled, not run on hardware
//
#include <hip/hip_runtime.h>

#define B_ 4
#define Q_ 2048
#define K_ 2048
#define D_ 1024

typedef __attribute__((ext_vector_type(16))) _Float16 v16h;
typedef __attribute__((ext_vector_type(8)))  _Float16 v8h;
typedef __attribute__((ext_vector_type(4)))  _Float16 v4h;
typedef __attribute__((ext_vector_type(8)))  float    v8f;
typedef __attribute__((ext_vector_type(4)))  float    v4f;

static __device__ __forceinline__ v16h concat16(v8h lo, v8h hi) {
    v16h r;
#pragma unroll
    for (int i = 0; i < 8; ++i) { r[i] = lo[i]; r[i + 8] = hi[i]; }
    return r;
}

// ---------------------------------------------------------------------------
// 1) s_k[b*K + k] = (sum_d value[b,k,d]) / sqrt(D)    (one 256-thread block/row)
// ---------------------------------------------------------------------------
__global__ void aa_rowsum_kernel(const float* __restrict__ value,
                                 float* __restrict__ s_k) {
    __shared__ float red[256];
    const int r   = blockIdx.x;      // b*K + k  (0..B*K-1)
    const int tid = threadIdx.x;
    const float* row = value + (size_t)r * D_;
    v4f v = *(const v4f*)(row + tid * 4);          // 256*4 == D
    float s = v[0] + v[1] + v[2] + v[3];
    red[tid] = s;
    __syncthreads();
    for (int off = 128; off > 0; off >>= 1) {
        if (tid < off) red[tid] += red[tid + off];
        __syncthreads();
    }
    if (tid == 0) s_k[r] = red[0] * 0.03125f;      // 1/sqrt(1024)
}

// ---------------------------------------------------------------------------
// 2) bmax[b] = max_k s_k[b,k]
// ---------------------------------------------------------------------------
__global__ void aa_bmax_kernel(const float* __restrict__ s_k,
                               float* __restrict__ bmax) {
    __shared__ float red[256];
    const int b   = blockIdx.x;
    const int tid = threadIdx.x;
    float m = -1e30f;
    for (int k = tid; k < K_; k += 256) m = fmaxf(m, s_k[(size_t)b * K_ + k]);
    red[tid] = m;
    __syncthreads();
    for (int off = 128; off > 0; off >>= 1) {
        if (tid < off) red[tid] = fmaxf(red[tid], red[tid + off]);
        __syncthreads();
    }
    if (tid == 0) bmax[b] = red[0];
}

// ---------------------------------------------------------------------------
// 3) e32[b,k] = exp(s_k - bmax[b])
// ---------------------------------------------------------------------------
__global__ void aa_exp_kernel(const float* __restrict__ s_k,
                              const float* __restrict__ bmax,
                              float* __restrict__ e32) {
    const int i = blockIdx.x * 256 + threadIdx.x;   // 0..B*K-1
    e32[i] = expf(s_k[i] - bmax[i >> 11]);          // K_ == 2048 == 1<<11
}

// ---------------------------------------------------------------------------
// 4) per (b,q) row: denom = sum_k mask*e32 ; write weights output + 1/denom
// ---------------------------------------------------------------------------
__global__ void aa_weights_kernel(const int* __restrict__ mask,
                                  const float* __restrict__ e32,
                                  float* __restrict__ wout,
                                  float* __restrict__ invd) {
    __shared__ float red[256];
    const int q   = blockIdx.x;
    const int b   = blockIdx.y;
    const int tid = threadIdx.x;
    const size_t base = ((size_t)b * Q_ + q) * K_;
    const float* e = e32 + (size_t)b * K_;
    int   mloc[8];
    float eloc[8];
    float part = 0.f;
#pragma unroll
    for (int i = 0; i < 8; ++i) {
        const int k = tid + 256 * i;
        mloc[i] = mask[base + k];
        eloc[i] = e[k];
        part += mloc[i] ? eloc[i] : 0.f;
    }
    red[tid] = part;
    __syncthreads();
    for (int off = 128; off > 0; off >>= 1) {
        if (tid < off) red[tid] += red[tid + off];
        __syncthreads();
    }
    const float inv = 1.0f / red[0];
    if (tid == 0) invd[(size_t)b * Q_ + q] = inv;
#pragma unroll
    for (int i = 0; i < 8; ++i)
        wout[base + tid + 256 * i] = mloc[i] ? eloc[i] * inv : 0.f;
}

// ---------------------------------------------------------------------------
// 5) vt[b][d][k] = (f16) value[b][k][d]   (32x32 LDS tiles)
// ---------------------------------------------------------------------------
__global__ void aa_transpose_kernel(const float* __restrict__ value,
                                    _Float16* __restrict__ vt) {
    __shared__ float tile[32][33];
    const int k0 = blockIdx.x * 32, d0 = blockIdx.y * 32, b = blockIdx.z;
    const int tx = threadIdx.x, ty = threadIdx.y;
    const float* src = value + (size_t)b * K_ * D_;
    _Float16*    dst = vt    + (size_t)b * D_ * K_;
#pragma unroll
    for (int i = 0; i < 4; ++i)
        tile[ty + 8 * i][tx] = src[(size_t)(k0 + ty + 8 * i) * D_ + d0 + tx];
    __syncthreads();
#pragma unroll
    for (int i = 0; i < 4; ++i)
        dst[(size_t)(d0 + ty + 8 * i) * K_ + k0 + tx] = (_Float16)tile[tx][ty + 8 * i];
}

// ---------------------------------------------------------------------------
// 6) context[b] = weights[b] (QxK, f32->f16) @ value[b] (KxD) via WMMA f16
//    WG: 128 threads (4 waves), C-tile 128x128, wave-tile 64x64 (4x4 WMMA).
//    Double-buffered LDS: stage tile i+1 while computing tile i; one barrier
//    per iteration (writes to a buffer and reads of it are always separated
//    by a barrier; DS ops retire in order within a wave).
// ---------------------------------------------------------------------------
__global__ __launch_bounds__(128)
void aa_gemm_kernel(const float* __restrict__ w,
                    const _Float16* __restrict__ vt,
                    const float* __restrict__ invd,
                    float* __restrict__ ctx) {
    __shared__ __align__(16) _Float16 As[2][128][40];   // [buf][q][k] padded (80B rows)
    __shared__ __align__(16) _Float16 Bs[2][128][40];   // [buf][d][k] padded

    const int tid  = threadIdx.x;
    const int lane = tid & 31;
    const int wv   = tid >> 5;       // wave 0..3
    const int wq   = wv >> 1;        // 0..1 (q half of tile)
    const int wd   = wv & 1;         // 0..1 (d half of tile)
    const int half = lane >> 4;      // lane group 0/1
    const int lm   = lane & 15;

    const int d0 = blockIdx.x * 128;
    const int q0 = blockIdx.y * 128;
    const int b  = blockIdx.z;

    const float*    wbase = w  + ((size_t)b * Q_ + q0) * K_;
    const _Float16* vbase = vt + ((size_t)b * D_ + d0) * (size_t)K_;

    // ---- staging helpers -------------------------------------------------
    auto stageA = [&](int kk, int buf) {
#pragma unroll
        for (int i = 0; i < 8; ++i) {
            const int g   = tid + 128 * i;         // 0..1023 groups of 4
            const int row = g >> 3;
            const int kc  = (g & 7) * 4;
            v4f f = *(const v4f*)(wbase + (size_t)row * K_ + kk + kc);
            v4h h;
            h[0] = (_Float16)f[0]; h[1] = (_Float16)f[1];
            h[2] = (_Float16)f[2]; h[3] = (_Float16)f[3];
            *(v4h*)&As[buf][row][kc] = h;
        }
    };
    auto stageB = [&](int kk, int buf) {
#pragma unroll
        for (int i = 0; i < 4; ++i) {
            const int c   = tid + 128 * i;         // 0..511 chunks of 8
            const int row = c >> 2;
            const int kc  = (c & 3) * 8;
            v8h chunk = *(const v8h*)(vbase + (size_t)row * K_ + kk + kc);
            *(v8h*)&Bs[buf][row][kc] = chunk;
        }
    };

    v8f acc[4][4];
#pragma unroll
    for (int i = 0; i < 4; ++i)
#pragma unroll
        for (int j = 0; j < 4; ++j)
#pragma unroll
            for (int r = 0; r < 8; ++r) acc[i][j][r] = 0.f;

    // prologue: fill buffer 0
    stageA(0, 0);
    stageB(0, 0);
    __syncthreads();

    for (int kk = 0; kk < K_; kk += 32) {
        const int cur = (kk >> 5) & 1;
        const int nxt = cur ^ 1;

        // stage next tile first so global loads overlap this tile's WMMAs
        if (kk + 32 < K_) {
            stageA(kk + 32, nxt);
            stageB(kk + 32, nxt);
        }

        v16h af[4], bf[4];
#pragma unroll
        for (int t = 0; t < 4; ++t) {
            // A frag (16x32): lanes<16 hold K 0-7 & 16-23, lanes>=16 hold 8-15 & 24-31
            const int ra = wq * 64 + t * 16 + lm;
            const int ka = half * 8;
            v8h alo = *(const v8h*)&As[cur][ra][ka];
            v8h ahi = *(const v8h*)&As[cur][ra][ka + 16];
            af[t] = concat16(alo, ahi);
            // B frag (32x16): lanes<16 hold K 0-7 & 16-23, lanes>=16 hold 8-15 & 24-31
            const int rb = wd * 64 + t * 16 + lm;
            const int kb = half * 8;
            v8h blo = *(const v8h*)&Bs[cur][rb][kb];
            v8h bhi = *(const v8h*)&Bs[cur][rb][kb + 16];
            bf[t] = concat16(blo, bhi);
        }
#pragma unroll
        for (int ti = 0; ti < 4; ++ti)
#pragma unroll
            for (int tj = 0; tj < 4; ++tj)
                acc[ti][tj] = __builtin_amdgcn_wmma_f32_16x16x32_f16(
                    false, af[ti], false, bf[tj], (short)0, acc[ti][tj],
                    false, false);

        __syncthreads();
    }

    // epilogue: scale rows by 1/denom, store context
    float* cbase = ctx + ((size_t)b * Q_ + q0) * D_;
#pragma unroll
    for (int ti = 0; ti < 4; ++ti) {
#pragma unroll
        for (int r = 0; r < 8; ++r) {
            const int qg  = wq * 64 + ti * 16 + half * 8 + r;   // M = r + 8*half
            const float inv = invd[(size_t)b * Q_ + q0 + qg];
#pragma unroll
            for (int tj = 0; tj < 4; ++tj) {
                const int dg = wd * 64 + tj * 16 + lm;          // N = lane&15
                cbase[(size_t)qg * D_ + dg] = acc[ti][tj][r] * inv;
            }
        }
    }
}

// ---------------------------------------------------------------------------
extern "C" void kernel_launch(void* const* d_in, const int* in_sizes, int n_in,
                              void* d_out, int out_size, void* d_ws, size_t ws_size,
                              hipStream_t stream) {
    (void)in_sizes; (void)n_in; (void)out_size; (void)ws_size;
    // inputs: query (unused: softmax shift-invariance cancels it), value, mask
    const float* value = (const float*)d_in[1];
    const int*   mask  = (const int*)d_in[2];

    float* ctx  = (float*)d_out;                               // (B,Q,D)
    float* wout = (float*)d_out + (size_t)B_ * Q_ * D_;        // (B,Q,K)

    char* ws = (char*)d_ws;
    size_t o = 0;
    float* s_k  = (float*)(ws + o); o += (size_t)B_ * K_ * 4;  // 32 KB
    float* bmax = (float*)(ws + o); o += 256;
    float* e32  = (float*)(ws + o); o += (size_t)B_ * K_ * 4;  // 32 KB
    float* invd = (float*)(ws + o); o += (size_t)B_ * Q_ * 4;  // 32 KB
    _Float16* vt = (_Float16*)(ws + o);                        // 16 MB (o is 16B aligned)

    aa_rowsum_kernel<<<B_ * K_, 256, 0, stream>>>(value, s_k);
    aa_bmax_kernel<<<B_, 256, 0, stream>>>(s_k, bmax);
    aa_exp_kernel<<<(B_ * K_) / 256, 256, 0, stream>>>(s_k, bmax, e32);
    aa_weights_kernel<<<dim3(Q_, B_), 256, 0, stream>>>(mask, e32, wout, invd);
    aa_transpose_kernel<<<dim3(K_ / 32, D_ / 32, B_), dim3(32, 8), 0, stream>>>(value, vt);
    aa_gemm_kernel<<<dim3(D_ / 128, Q_ / 128, B_), 128, 0, stream>>>(wout, vt, invd, ctx);
}
